// ManyCodebooksVQBottleneck_56410100465695
// MI455X (gfx1250) — matrix operand order repaired and checked by reading, not compiled, not run on hardware
//
#include <hip/hip_runtime.h>
#include <hip/hip_bf16.h>
#include <math.h>

typedef __attribute__((ext_vector_type(16))) _Float16 v16h;
typedef __attribute__((ext_vector_type(8)))  _Float16 v8h;
typedef __attribute__((ext_vector_type(8)))  float    v8f;

#define NCB    4
#define MCODES 512
#define DDIM   128
#define HW     1024           // H*W
#define TTOK   32768          // B*H*W
#define OUT_ELEMS (NCB * TTOK * DDIM)   // 16777216

// workspace layout (bytes)
#define WS_EH     0u                    // f16 [4][512][128]  = 524288 B
#define WS_ENORM  524288u               // f32 [4][512]       =   8192 B
#define WS_IDX    532480u               // i32 [4][32768]     = 524288 B
#define WS_COUNTS 1056768u              // i32 [2048]         =   8192 B
#define WS_ACCUM  1064960u              // f32 [1]

// ---------------------------------------------------------------------------
// CDNA5 async global -> LDS copy (16 B per lane), tracked by ASYNCcnt.
// ---------------------------------------------------------------------------
__device__ __forceinline__ void async_copy16(void* lds_dst, const void* gsrc) {
    unsigned int l = (unsigned int)(unsigned long long)(uintptr_t)lds_dst; // low 32b = LDS offset
    unsigned long long g = (unsigned long long)(uintptr_t)gsrc;
    asm volatile("global_load_async_to_lds_b128 %0, %1, off"
                 :: "v"(l), "v"(g) : "memory");
}
__device__ __forceinline__ void wait_async0() {
    asm volatile("s_wait_asynccnt 0x0" ::: "memory");
}

// ---------------------------------------------------------------------------
// Kernel 1: convert embedding fp32 -> f16, compute ||e||^2 per code,
//           zero the count histogram and loss accumulator.
// grid = 2048 blocks (n*512+m), block = 128 threads (d)
// ---------------------------------------------------------------------------
__global__ void vq_prep(const float* __restrict__ emb, _Float16* __restrict__ eh,
                        float* __restrict__ enorm, int* __restrict__ counts,
                        float* __restrict__ accum) {
    const int bid = blockIdx.x;        // n*512 + m
    const int d   = threadIdx.x;       // 0..127
    const float v = emb[(size_t)bid * DDIM + d];
    eh[(size_t)bid * DDIM + d] = (_Float16)v;
    __shared__ float red[128];
    red[d] = v * v;
    __syncthreads();
    for (int s = 64; s > 0; s >>= 1) {
        if (d < s) red[d] += red[d + s];
        __syncthreads();
    }
    if (d == 0) {
        enorm[bid]  = red[0];
        counts[bid] = 0;
        if (bid == 0) *accum = 0.0f;
    }
}

// ---------------------------------------------------------------------------
// Kernel 2: distance GEMM via WMMA + argmin, with double-buffered async
// global->LDS staging of the codebook tiles.
//
// Workgroup = 8 waves, one codebook n, 16 token tiles (2 per wave).
//   A (SRC0) = 16 codes x 32 K (f16 from LDS tile)
//   B (SRC1) = 32 K x 16 tokens x2 (f16 converted from x, held in VGPRs)
//   D[code][token] accumulated over 4 K-chunks (D = 128).
// C/D layout: lane l, VGPR v holds D[v + 8*(l>>4)][l&15].
// grid = 512 blocks (4 codebooks x 128 groups), block = 256 threads
// ---------------------------------------------------------------------------
__global__ void __launch_bounds__(256)
vq_argmin(const float* __restrict__ x, const _Float16* __restrict__ eh,
          const float* __restrict__ enorm, int* __restrict__ idx,
          int* __restrict__ counts) {
    __shared__ _Float16 sA[2][16 * DDIM];   // 2 x 4 KB code tiles
    __shared__ float    sEn[MCODES];        // 2 KB ||e||^2

    const int tid  = threadIdx.x;
    const int lane = tid & 31;
    const int wid  = tid >> 5;
    const int n    = blockIdx.x >> 7;          // 0..3
    const int grp  = blockIdx.x & 127;         // 128 groups of 16 tiles
    const int col  = lane & 15;
    const int half = lane >> 4;

    const _Float16* ehn = eh + (size_t)n * MCODES * DDIM;

    // ---- stage ||e||^2 (512 f32 = 128 chunks) and first A tile (ct=0) ----
    if (tid < 128)
        async_copy16((char*)sEn + tid * 16, enorm + n * MCODES + tid * 4);
    async_copy16((char*)sA[0] + tid * 16, ehn + tid * 8);   // 256 x 16 B = 4 KB

    // ---- load two B (token) tiles while the async copies are in flight ----
    const int ta = grp * 16 + wid * 2;      // token-tile indices (0..2047)
    const int tb = ta + 1;
    v16h bmatA[4], bmatB[4];
#pragma unroll
    for (int t = 0; t < 2; ++t) {
        const int tt   = t ? tb : ta;
        const int t0   = tt << 4;
        const int b    = t0 >> 10;
        const int pos0 = t0 & 1023;
        const float* xb = x + ((size_t)(b * NCB + n) * DDIM) * HW + pos0 + col;
        v16h* bm = t ? bmatB : bmatA;
#pragma unroll
        for (int c = 0; c < 4; ++c) {
            // element e of chunk c: token = t0+col, d = c*32 + 16*half + e
            const float* xc = xb + (size_t)(c * 32 + 16 * half) * HW;
#pragma unroll
            for (int e = 0; e < 16; ++e)
                bm[c][e] = (_Float16)xc[(size_t)e * HW];
        }
    }

    wait_async0();
    __syncthreads();

    float best0 = 3.4e38f, best1 = 3.4e38f;
    int   bi0 = 0, bi1 = 0;

    for (int ct = 0; ct < 32; ++ct) {       // 32 code tiles of 16
        const int cur = ct & 1;
        if (ct < 31)                         // prefetch next tile into other buf
            async_copy16((char*)sA[cur ^ 1] + tid * 16,
                         ehn + (size_t)(ct + 1) * (16 * DDIM) + tid * 8);

        v8f acc0 = {}, acc1 = {};
        // A layout: row = col; half h element e -> k = 16*h + e
        const _Float16* abase = sA[cur] + col * DDIM + half * 16;
#pragma unroll
        for (int c = 0; c < 4; ++c) {
            v8h r0 = *(const v8h*)(abase + c * 32);        // k = 16h..16h+7
            v8h r1 = *(const v8h*)(abase + c * 32 + 8);    // k = 16h+8..16h+15
            v16h a;
#pragma unroll
            for (int i = 0; i < 8; ++i) { a[i] = r0[i]; a[i + 8] = r1[i]; }
            acc0 = __builtin_amdgcn_wmma_f32_16x16x32_f16(
                false, a, false, bmatA[c], (short)0, acc0, false, false);
            acc1 = __builtin_amdgcn_wmma_f32_16x16x32_f16(
                false, a, false, bmatB[c], (short)0, acc1, false, false);
        }

        // distances for 8 codes per lane: code = ct*16 + v + 8*half
        const float4* ep = (const float4*)(sEn + ct * 16 + 8 * half);
        float4 e0 = ep[0], e1 = ep[1];
        float en8[8] = {e0.x, e0.y, e0.z, e0.w, e1.x, e1.y, e1.z, e1.w};
#pragma unroll
        for (int v = 0; v < 8; ++v) {
            int   code = ct * 16 + v + 8 * half;
            float d0 = en8[v] - 2.0f * acc0[v];
            bool  l0 = d0 < best0;                // strict <: first-index ties
            bi0   = l0 ? code : bi0;
            best0 = l0 ? d0 : best0;
            float d1 = en8[v] - 2.0f * acc1[v];
            bool  l1 = d1 < best1;
            bi1   = l1 ? code : bi1;
            best1 = l1 ? d1 : best1;
        }

        wait_async0();       // own async writes landed
        __syncthreads();     // everyone's writes visible, buf free for reuse
    }

    // merge lane halves (same token, disjoint code subsets), lower-index ties
    {
        float ob = __shfl_xor(best0, 16, 32);
        int   oi = __shfl_xor(bi0, 16, 32);
        if (ob < best0 || (ob == best0 && oi < bi0)) { best0 = ob; bi0 = oi; }
        ob = __shfl_xor(best1, 16, 32);
        oi = __shfl_xor(bi1, 16, 32);
        if (ob < best1 || (ob == best1 && oi < bi1)) { best1 = ob; bi1 = oi; }
    }
    if (half == 0) {
        idx[(size_t)n * TTOK + (ta << 4) + col] = bi0;
        atomicAdd(&counts[n * MCODES + bi0], 1);
        idx[(size_t)n * TTOK + (tb << 4) + col] = bi1;
        atomicAdd(&counts[n * MCODES + bi1], 1);
    }
}

// ---------------------------------------------------------------------------
// Kernel 3: gather quantized vectors (fp32 embedding), write output in
// (B,C,H,W) order, accumulate sum((x-q)^2).
// grid = 8192 blocks (n, 16-token tile), block = 256 threads
// ---------------------------------------------------------------------------
__global__ void __launch_bounds__(256)
vq_gather(const float* __restrict__ x, const float* __restrict__ emb,
          const int* __restrict__ idx, float* __restrict__ out,
          float* __restrict__ accum) {
    const int n    = blockIdx.x >> 11;     // 0..3
    const int tile = blockIdx.x & 2047;
    const int t0   = tile << 4;
    const int b    = t0 >> 10;
    const int pos0 = t0 & 1023;
    const int tid  = threadIdx.x;

    __shared__ int   sidx[16];
    __shared__ float red[256];
    if (tid < 16) sidx[tid] = idx[(size_t)n * TTOK + t0 + tid];
    __syncthreads();

    const size_t base = ((size_t)(b * NCB + n) * DDIM) * HW + pos0;
    float local = 0.0f;
#pragma unroll
    for (int it = 0; it < 8; ++it) {
        int elem = it * 256 + tid;          // 0..2047 over (d, token)
        int d    = elem >> 4;
        int ts   = elem & 15;
        float q  = emb[((size_t)n * MCODES + sidx[ts]) * DDIM + d];
        size_t a = base + (size_t)d * HW + ts;
        float xv = x[a];
        out[a]   = q;                       // straight-through forward == q
        float df = xv - q;
        local += df * df;
    }
    red[tid] = local;
    __syncthreads();
    for (int s = 128; s > 0; s >>= 1) {
        if (tid < s) red[tid] += red[tid + s];
        __syncthreads();
    }
    if (tid == 0) atomicAdd(accum, red[0]);
}

// ---------------------------------------------------------------------------
// Kernel 4: entropy / perplexity / loss scalars.  1 block x 256 threads.
// ---------------------------------------------------------------------------
__global__ void vq_finalize(const int* __restrict__ counts,
                            const float* __restrict__ accum,
                            float* __restrict__ outScalars) {
    __shared__ float red[256];
    __shared__ float ent[NCB];
    const int tid = threadIdx.x;
    for (int n = 0; n < NCB; ++n) {
        float p = 0.0f;
#pragma unroll
        for (int j = 0; j < 2; ++j) {
            int m = tid + j * 256;
            float avg = (float)counts[n * MCODES + m] / (float)TTOK;
            p += avg * logf(avg + 1e-10f);
        }
        red[tid] = p;
        __syncthreads();
        for (int s = 128; s > 0; s >>= 1) {
            if (tid < s) red[tid] += red[tid + s];
            __syncthreads();
        }
        if (tid == 0) ent[n] = -red[0];
        __syncthreads();
    }
    if (tid == 0) {
        float pplx = 0.0f, pfix = 0.0f;
        for (int n = 0; n < NCB; ++n) { pplx += expf(ent[n]); pfix += ent[n]; }
        // vq_loss = (1.0 + 0.25) * mean((x-q)^2) : both terms equal in fwd
        outScalars[0] = (*accum) * (1.25f / (float)OUT_ELEMS);
        outScalars[1] = pplx;
        outScalars[2] = pfix;
    }
}

// ---------------------------------------------------------------------------
extern "C" void kernel_launch(void* const* d_in, const int* in_sizes, int n_in,
                              void* d_out, int out_size, void* d_ws, size_t ws_size,
                              hipStream_t stream) {
    const float* x   = (const float*)d_in[0];   // (32,512,32,32) f32
    const float* emb = (const float*)d_in[1];   // (4,512,128) f32
    float* out = (float*)d_out;                 // 16777216 + 3 scalars

    char* ws = (char*)d_ws;
    _Float16* eh     = (_Float16*)(ws + WS_EH);
    float*    enorm  = (float*)(ws + WS_ENORM);
    int*      idx    = (int*)(ws + WS_IDX);
    int*      counts = (int*)(ws + WS_COUNTS);
    float*    accum  = (float*)(ws + WS_ACCUM);

    vq_prep    <<<NCB * MCODES, 128, 0, stream>>>(emb, eh, enorm, counts, accum);
    vq_argmin  <<<512,          256, 0, stream>>>(x, eh, enorm, idx, counts);
    vq_gather  <<<8192,         256, 0, stream>>>(x, emb, idx, out, accum);
    vq_finalize<<<1,            256, 0, stream>>>(counts, accum, out + OUT_ELEMS);
}
